// GPT2Attention_70300024701797
// MI455X (gfx1250) — matrix operand order repaired and checked by reading, not compiled, not run on hardware
//
#include <hip/hip_runtime.h>

// ---------------------------------------------------------------------------
// CDNA5 (gfx1250) GPT-2 attention forward, bf16 WMMA pipeline.
//   x[2,2048,1024] f32 -> out[2,2048,1024] f32
// Stages: cvt/transpose -> QKV gemm (wmma) -> flash attention (wmma) -> proj gemm (wmma)
// ---------------------------------------------------------------------------

typedef __bf16 bf16;
typedef __attribute__((ext_vector_type(16))) __bf16 v16bf;
typedef __attribute__((ext_vector_type(8)))  __bf16 v8bf;
typedef __attribute__((ext_vector_type(8)))  float  v8f;

#define EMB   1024
#define SEQ   2048
#define BATCH 2
#define HEADS 16
#define HEAD  64
#define ROWS  (BATCH * SEQ)        // 4096
#define N3E   (3 * EMB)            // 3072

__device__ __forceinline__ v8f wmma_bf16(v16bf a, v16bf b, v8f c) {
  // D = A(16x32 bf16) x B(32x16 bf16) + C(16x16 f32)
  return __builtin_amdgcn_wmma_f32_16x16x32_bf16(
      /*neg_a=*/false, a, /*neg_b=*/false, b,
      /*c_mod=*/(short)0, c, /*reuse_a=*/false, /*reuse_b=*/false);
}

// A-matrix 16x32 fragment, row-major source (rows of length `ld` bf16).
// ISA layout: lane (m + 16h) holds row m, elements e<8 -> K=8h+e, e>=8 -> K=16+8h+(e-8).
__device__ __forceinline__ v16bf load_A_frag(const bf16* __restrict__ base,
                                             int ld, int row0, int k0) {
  const int lane = threadIdx.x & 31;
  const int m = lane & 15, h = lane >> 4;
  const bf16* p = base + (size_t)(row0 + m) * ld + k0 + 8 * h;
  v8bf lo = *(const v8bf*)(p);
  v8bf hi = *(const v8bf*)(p + 16);
  v16bf r;
#pragma unroll
  for (int i = 0; i < 8; ++i) { r[i] = lo[i]; r[8 + i] = hi[i]; }
  return r;
}

// B-matrix 32x16 fragment from a TRANSPOSED source Bt[n][k] (row length `ld`).
// ISA layout: lane (n + 16h) holds column n, elements e<8 -> K=8h+e, e>=8 -> K=16+8h+(e-8).
__device__ __forceinline__ v16bf load_Bt_frag(const bf16* __restrict__ base,
                                              int ld, int n0, int k0) {
  const int lane = threadIdx.x & 31;
  const int n = lane & 15, h = lane >> 4;
  const bf16* p = base + (size_t)(n0 + n) * ld + k0 + 8 * h;
  v8bf lo = *(const v8bf*)(p);
  v8bf hi = *(const v8bf*)(p + 16);
  v16bf r;
#pragma unroll
  for (int i = 0; i < 8; ++i) { r[i] = lo[i]; r[8 + i] = hi[i]; }
  return r;
}

// ------------------------------ converts -----------------------------------
__global__ void cvt_f32_bf16(const float* __restrict__ src, bf16* __restrict__ dst, int n) {
  int i = blockIdx.x * blockDim.x + threadIdx.x;
  if (i < n) dst[i] = (bf16)src[i];
}

// W[K,N] f32 -> Wt[N,K] bf16 (coalesced read, scattered write; one-time, tiny traffic)
__global__ void transpose_f32_bf16(const float* __restrict__ src, bf16* __restrict__ dst,
                                   int K, int N) {
  int i = blockIdx.x * blockDim.x + threadIdx.x;
  if (i < K * N) {
    int k = i / N, n = i % N;
    dst[(size_t)n * K + k] = (bf16)src[i];
  }
}

// ------------------------------ QKV GEMM -----------------------------------
// C[ROWS, 3E] = Xbf[ROWS, E] @ WaT^T + b ; scatter into Q/K [b,h,s,d] and Vt [b,h,d,s].
// One wave computes a 32(M) x 64(N) tile.
__global__ void qkv_gemm(const bf16* __restrict__ X, const bf16* __restrict__ WaT,
                         const float* __restrict__ bias,
                         bf16* __restrict__ Q, bf16* __restrict__ Kb,
                         bf16* __restrict__ Vt) {
  const int lane = threadIdx.x & 31;
  const int n0 = blockIdx.x * 64;
  const int m0 = blockIdx.y * 32;
  v8f acc[2][4];
#pragma unroll
  for (int i = 0; i < 2; ++i)
#pragma unroll
    for (int j = 0; j < 4; ++j) acc[i][j] = (v8f){};

  for (int kk = 0; kk < EMB; kk += 32) {
    v16bf a0 = load_A_frag(X, EMB, m0, kk);
    v16bf a1 = load_A_frag(X, EMB, m0 + 16, kk);
#pragma unroll
    for (int ni = 0; ni < 4; ++ni) {
      v16bf b = load_Bt_frag(WaT, EMB, n0 + 16 * ni, kk);
      acc[0][ni] = wmma_bf16(a0, b, acc[0][ni]);
      acc[1][ni] = wmma_bf16(a1, b, acc[1][ni]);
    }
  }

  const int nl = lane & 15, h = lane >> 4;
#pragma unroll
  for (int ni = 0; ni < 4; ++ni) {
    const int col = n0 + ni * 16 + nl;
    const float bv = bias[col];
    const int part = col >> 10;            // 0=Q 1=K 2=V
    const int rem = col & (EMB - 1);
    const int hj = rem >> 6, dj = rem & 63;
#pragma unroll
    for (int mi = 0; mi < 2; ++mi) {
#pragma unroll
      for (int r = 0; r < 8; ++r) {
        const int row = m0 + mi * 16 + 8 * h + r;        // global row in [0,ROWS)
        const int bb = row >> 11, s = row & (SEQ - 1);
        const bf16 v = (bf16)(acc[mi][ni][r] + bv);
        const size_t hd = (size_t)(bb * HEADS + hj);
        if (part == 0)      Q [(hd * SEQ + s) * HEAD + dj] = v;
        else if (part == 1) Kb[(hd * SEQ + s) * HEAD + dj] = v;
        else                Vt[(hd * HEAD + dj) * SEQ + s] = v;
      }
    }
  }
}

// ---------------------------- flash attention ------------------------------
// Per wave: one (b,h,q-tile of 16). Computes S^T = K @ Q^T so softmax stats are
// per-lane; the C-layout of S^T coincides with the A-fragment layout of P for
// O = P @ V (no cross-lane movement).
__global__ void attn_kernel(const bf16* __restrict__ Q, const bf16* __restrict__ Kb,
                            const bf16* __restrict__ Vt, bf16* __restrict__ O) {
  const int lane = threadIdx.x & 31;
  const int q0 = blockIdx.x * 16;
  const int h  = blockIdx.y;
  const int b  = blockIdx.z;
  const size_t hoff = (size_t)(b * HEADS + h) * SEQ * HEAD;
  const bf16* Qh = Q  + hoff;
  const bf16* Kh = Kb + hoff;
  const bf16* Vh = Vt + hoff;   // laid out [d][s]

  // Q^T as B fragments (loop invariant): lane n holds column q0+n, K-dim = d.
  const v16bf qb0 = load_Bt_frag(Qh, HEAD, q0, 0);
  const v16bf qb1 = load_Bt_frag(Qh, HEAD, q0, 32);

  v8f o[4];
#pragma unroll
  for (int i = 0; i < 4; ++i) o[i] = (v8f){};
  float m_run = -__builtin_inff();
  float l_run = 0.f;
  const int hh = lane >> 4;
  const v8f zero = (v8f){};

  for (int k0 = 0; k0 < SEQ; k0 += 32) {
    // S^T tile rows k0..k0+31, cols q0..q0+15 : 4 WMMAs over d=64
    v16bf ka0 = load_A_frag(Kh, HEAD, k0,      0);
    v16bf ka1 = load_A_frag(Kh, HEAD, k0,      32);
    v16bf ka2 = load_A_frag(Kh, HEAD, k0 + 16, 0);
    v16bf ka3 = load_A_frag(Kh, HEAD, k0 + 16, 32);
    v8f d1 = wmma_bf16(ka0, qb0, zero);
    d1 = wmma_bf16(ka1, qb1, d1);
    v8f d2 = wmma_bf16(ka2, qb0, zero);
    d2 = wmma_bf16(ka3, qb1, d2);

    // softmax (scale 1/sqrt(64) = 0.125)
    float s1[8], s2[8];
    float tmax = -__builtin_inff();
#pragma unroll
    for (int r = 0; r < 8; ++r) {
      s1[r] = d1[r] * 0.125f;
      s2[r] = d2[r] * 0.125f;
      tmax = fmaxf(tmax, fmaxf(s1[r], s2[r]));
    }
    tmax = fmaxf(tmax, __shfl_xor(tmax, 16, 32));
    const float m_new = fmaxf(m_run, tmax);
    const float alpha = __expf(m_run - m_new);
    float p1[8], p2[8], ls = 0.f;
#pragma unroll
    for (int r = 0; r < 8; ++r) {
      p1[r] = __expf(s1[r] - m_new);
      p2[r] = __expf(s2[r] - m_new);
      ls += p1[r] + p2[r];
    }
    ls += __shfl_xor(ls, 16, 32);
    l_run = l_run * alpha + ls;
    m_run = m_new;

    // P A-fragment: element e<8 from d1[e], e>=8 from d2[e-8] (layout identity).
    v16bf pa;
#pragma unroll
    for (int r = 0; r < 8; ++r) { pa[r] = (bf16)p1[r]; pa[8 + r] = (bf16)p2[r]; }

    // per-row rescale factors for the O accumulator (C layout: VGPR r -> q=8*hh+r)
    float arow[8];
#pragma unroll
    for (int r = 0; r < 8; ++r) arow[r] = __shfl(alpha, 8 * hh + r, 32);

#pragma unroll
    for (int ni = 0; ni < 4; ++ni) {
      v16bf vb = load_Bt_frag(Vh, SEQ, ni * 16, k0);  // V[k, d0+n] = Vt[d0+n][k]
      v8f t = o[ni];
#pragma unroll
      for (int r = 0; r < 8; ++r) t[r] *= arow[r];
      o[ni] = wmma_bf16(pa, vb, t);
    }
  }

  const float rl = 1.f / l_run;
  float rrow[8];
#pragma unroll
  for (int r = 0; r < 8; ++r) rrow[r] = __shfl(rl, 8 * hh + r, 32);
  const int nl = lane & 15;
#pragma unroll
  for (int ni = 0; ni < 4; ++ni) {
#pragma unroll
    for (int r = 0; r < 8; ++r) {
      const int qrow = q0 + 8 * hh + r;
      O[(size_t)(b * SEQ + qrow) * EMB + h * HEAD + ni * 16 + nl] =
          (bf16)(o[ni][r] * rrow[r]);
    }
  }
}

// ------------------------------ proj GEMM ----------------------------------
__global__ void proj_gemm(const bf16* __restrict__ A, const bf16* __restrict__ WpT,
                          const float* __restrict__ bias, float* __restrict__ out) {
  const int lane = threadIdx.x & 31;
  const int n0 = blockIdx.x * 64;
  const int m0 = blockIdx.y * 32;
  v8f acc[2][4];
#pragma unroll
  for (int i = 0; i < 2; ++i)
#pragma unroll
    for (int j = 0; j < 4; ++j) acc[i][j] = (v8f){};

  for (int kk = 0; kk < EMB; kk += 32) {
    v16bf a0 = load_A_frag(A, EMB, m0, kk);
    v16bf a1 = load_A_frag(A, EMB, m0 + 16, kk);
#pragma unroll
    for (int ni = 0; ni < 4; ++ni) {
      v16bf bfr = load_Bt_frag(WpT, EMB, n0 + 16 * ni, kk);
      acc[0][ni] = wmma_bf16(a0, bfr, acc[0][ni]);
      acc[1][ni] = wmma_bf16(a1, bfr, acc[1][ni]);
    }
  }

  const int nl = lane & 15, h = lane >> 4;
#pragma unroll
  for (int ni = 0; ni < 4; ++ni) {
    const int col = n0 + ni * 16 + nl;
    const float bv = bias[col];
#pragma unroll
    for (int mi = 0; mi < 2; ++mi) {
#pragma unroll
      for (int r = 0; r < 8; ++r) {
        const int row = m0 + mi * 16 + 8 * h + r;
        out[(size_t)row * EMB + col] = acc[mi][ni][r] + bv;
      }
    }
  }
}

// ------------------------------ launcher -----------------------------------
extern "C" void kernel_launch(void* const* d_in, const int* in_sizes, int n_in,
                              void* d_out, int out_size, void* d_ws, size_t ws_size,
                              hipStream_t stream) {
  const float* x      = (const float*)d_in[0];   // [2,2048,1024]
  const float* w_attn = (const float*)d_in[1];   // [1024,3072]
  const float* b_attn = (const float*)d_in[2];   // [3072]
  const float* w_proj = (const float*)d_in[3];   // [1024,1024]
  const float* b_proj = (const float*)d_in[4];   // [1024]
  float* out = (float*)d_out;

  char* ws = (char*)d_ws;
  size_t off = 0;
  bf16* Xbf = (bf16*)(ws + off); off += (size_t)ROWS * EMB * 2;        // 8 MB
  bf16* WaT = (bf16*)(ws + off); off += (size_t)N3E  * EMB * 2;        // 6 MB
  bf16* WpT = (bf16*)(ws + off); off += (size_t)EMB  * EMB * 2;        // 2 MB
  bf16* Qb  = (bf16*)(ws + off); off += (size_t)ROWS * EMB * 2;        // 8 MB
  bf16* Kb  = (bf16*)(ws + off); off += (size_t)ROWS * EMB * 2;        // 8 MB
  bf16* Vt  = (bf16*)(ws + off); off += (size_t)ROWS * EMB * 2;        // 8 MB
  bf16* Ob  = (bf16*)(ws + off); off += (size_t)ROWS * EMB * 2;        // 8 MB

  const int NX = ROWS * EMB;
  cvt_f32_bf16<<<(NX + 255) / 256, 256, 0, stream>>>(x, Xbf, NX);
  transpose_f32_bf16<<<(EMB * N3E + 255) / 256, 256, 0, stream>>>(w_attn, WaT, EMB, N3E);
  transpose_f32_bf16<<<(EMB * EMB + 255) / 256, 256, 0, stream>>>(w_proj, WpT, EMB, EMB);

  qkv_gemm<<<dim3(N3E / 64, ROWS / 32), 32, 0, stream>>>(Xbf, WaT, b_attn, Qb, Kb, Vt);
  attn_kernel<<<dim3(SEQ / 16, HEADS, BATCH), 32, 0, stream>>>(Qb, Kb, Vt, Ob);
  proj_gemm<<<dim3(EMB / 64, ROWS / 32), 32, 0, stream>>>(Ob, WpT, b_proj, out);
}
